// SlotMixture_86998857547868
// MI455X (gfx1250) — compile-verified, matrix operand order checked by reading, not run on hardware
//
#include <hip/hip_runtime.h>
#include <hip/hip_bf16.h>
#include <math.h>

#define BB 4
#define NN 4096
#define DD 128
#define NS 16
#define HID 512
#define NTILES (NN/16)          // 256
#define EPSV 1e-10f
#define SCALE 0.08838834764831845f  // 128^-0.5

typedef __attribute__((ext_vector_type(16))) _Float16 v16h;
typedef __attribute__((ext_vector_type(8)))  float    v8f;

// K-offset pattern inside one 32-wide k-step for 16-bit WMMA A/B fragments
// (ISA 7.12.2: VGPR p<4 -> K=2p,2p+1 (+8 for lanes 16-31); p>=4 -> K=16+2(p-4),...)
__device__ __forceinline__ int kpat(int p, int hi) {
  return ((p < 4) ? (2 * p) : (16 + 2 * (p - 4))) + (hi ? 8 : 0);
}

// A-style fragment from row-major f16 matrix (lane%16 -> M/N row; contiguous pairs)
__device__ __forceinline__ v16h fragA_h(const _Float16* base, int ld, int row, int k0, int hi) {
  v16h f;
#pragma unroll
  for (int p = 0; p < 8; ++p) {
    int kb = k0 + kpat(p, hi);
    const _Float16* s = base + (size_t)row * ld + kb;
    f[2 * p] = s[0];
    f[2 * p + 1] = s[1];
  }
  return f;
}

// A-style fragment from row-major f32 matrix, converting to f16 on load
__device__ __forceinline__ v16h fragA_f32(const float* base, int ld, int row, int k0, int hi) {
  v16h f;
#pragma unroll
  for (int p = 0; p < 8; ++p) {
    int kb = k0 + kpat(p, hi);
    const float* s = base + (size_t)row * ld + kb;
    f[2 * p] = (_Float16)s[0];
    f[2 * p + 1] = (_Float16)s[1];
  }
  return f;
}

__device__ __forceinline__ v8f wmma_f16(v16h a, v16h b, v8f c) {
  return __builtin_amdgcn_wmma_f32_16x16x32_f16(false, a, false, b, (short)0, c, false, false);
}

// block-wide sum, blockDim.x up to 256, wave32
__device__ __forceinline__ float blk_sum(float v, float* sm) {
#pragma unroll
  for (int o = 16; o; o >>= 1) v += __shfl_xor(v, o, 32);
  int w = threadIdx.x >> 5, nw = blockDim.x >> 5;
  if ((threadIdx.x & 31) == 0) sm[w] = v;
  __syncthreads();
  if (threadIdx.x == 0) {
    float t = 0.f;
    for (int i = 0; i < nw; ++i) t += sm[i];
    sm[0] = t;
  }
  __syncthreads();
  float t = sm[0];
  __syncthreads();
  return t;
}

// ---------------- Stage 0: LayerNorm(inputs) -> fp16 ----------------
__global__ void k_ln_in(const float* __restrict__ x, const float* __restrict__ g,
                        const float* __restrict__ be, _Float16* __restrict__ xh) {
  __shared__ float sm[8];
  int row = blockIdx.x;          // b*n rows
  int t = threadIdx.x;           // 128
  float v = x[(size_t)row * DD + t];
  float mean = blk_sum(v, sm) * (1.f / DD);
  float dv = v - mean;
  float var = blk_sum(dv * dv, sm) * (1.f / DD);
  float y = dv * rsqrtf(var + 1e-5f) * g[t] + be[t];
  xh[(size_t)row * DD + t] = (_Float16)y;
}

// ---------------- Stage 1: k = x@Wk^T, v = x@Wv^T (WMMA) -------------
// one wave per 16-row M tile; stores k,k^2 row-major f16 and v TRANSPOSED
// (vT[b][d][n]) so the A@V GEMM's B fragments become contiguous loads.
__global__ void k_proj(const _Float16* __restrict__ xh, const float* __restrict__ Wk,
                       const float* __restrict__ Wv, _Float16* __restrict__ kh,
                       _Float16* __restrict__ k2h, _Float16* __restrict__ vt) {
  int lane = threadIdx.x & 31, wv = threadIdx.x >> 5;
  int mtile = blockIdx.x * 4 + wv;      // 1024 tiles over b*n rows
  int row0 = mtile * 16;
  int hi = lane >> 4, lr = lane & 15;
  v16h a[4];
#pragma unroll
  for (int s = 0; s < 4; ++s) a[s] = fragA_h(xh, DD, row0 + lr, s * 32, hi);
#pragma unroll
  for (int w = 0; w < 2; ++w) {
    const float* W = w ? Wv : Wk;
#pragma unroll
    for (int nt = 0; nt < 8; ++nt) {
      v8f acc = {};
#pragma unroll
      for (int s = 0; s < 4; ++s) {
        v16h bf = fragA_f32(W, DD, nt * 16 + lr, s * 32, hi);  // B = W^T => row-major load of W
        acc = wmma_f16(a[s], bf, acc);
      }
#pragma unroll
      for (int r = 0; r < 8; ++r) {
        int gm = row0 + r + hi * 8;          // global row over b*n
        float val = acc[r];
        if (w == 0) {
          size_t idx = (size_t)gm * DD + nt * 16 + lr;
          kh[idx] = (_Float16)val;
          k2h[idx] = (_Float16)(val * val);
        } else {
          int bb = gm / NN, j = gm % NN;     // transposed store: contiguous in j per lane
          vt[((size_t)bb * DD + nt * 16 + lr) * NN + j] = (_Float16)val;
        }
      }
    }
  }
}

// ------------- Per-iter A: slot LN + q projections -------------------
__global__ void k_slot_pre(const float* __restrict__ slots, const float* __restrict__ g_sl,
                           const float* __restrict__ be_sl, const float* __restrict__ Wq,
                           float* __restrict__ iv_f, float* __restrict__ qiv_f,
                           float* __restrict__ c_sum, float* __restrict__ mu_f) {
  __shared__ float sm[8];
  __shared__ float mu[DD], lg[DD];
  int row = blockIdx.x;          // b*ns = 64
  int t = threadIdx.x;           // 128
  const float* s = slots + (size_t)row * 2 * DD;
  float v0 = s[t], v1 = s[t + DD];
  float mean = blk_sum(v0 + v1, sm) * (1.f / (2 * DD));
  float d0 = v0 - mean, d1 = v1 - mean;
  float var = blk_sum(d0 * d0 + d1 * d1, sm) * (1.f / (2 * DD));
  float inv = rsqrtf(var + 1e-5f);
  mu[t] = d0 * inv * g_sl[t] + be_sl[t];
  lg[t] = d1 * inv * g_sl[t + DD] + be_sl[t + DD];
  __syncthreads();
  float qm = 0.f, ql = 0.f;
  const float* wr = Wq + t * DD;
  for (int c = 0; c < DD; ++c) { qm += mu[c] * wr[c]; ql += lg[c] * wr[c]; }
  float iv = __expf(-2.f * ql);
  iv_f[row * DD + t] = iv;
  qiv_f[row * DD + t] = qm * iv;
  mu_f[row * DD + t] = mu[t];
  float cc = blk_sum(qm * qm * iv, sm);
  if (t == 0) c_sum[row] = cc;
}

// ------------- Per-iter B: dots GEMM + slot softmax (fused) ----------
// dots = scale*(k2@iv - 2*k@(q*iv) + c).  C layout: lane%16 = slot (N),
// rows = points (M) => softmax over slots is a 16-lane xor reduction.
__global__ void k_dots(const _Float16* __restrict__ kh, const _Float16* __restrict__ k2h,
                       const float* __restrict__ iv_f, const float* __restrict__ qiv_f,
                       const float* __restrict__ c_sum, float* __restrict__ attn,
                       float* __restrict__ partials) {
  int lane = threadIdx.x & 31, wv = threadIdx.x >> 5;
  int bb = blockIdx.y;
  int mtile = blockIdx.x * 4 + wv;       // 0..255 per batch
  int hi = lane >> 4, lr = lane & 15;
  const _Float16* kb_ = kh + (size_t)bb * NN * DD;
  const _Float16* k2b = k2h + (size_t)bb * NN * DD;
  int row0 = mtile * 16;
  v8f g1 = {}, g2 = {};
#pragma unroll
  for (int s = 0; s < 4; ++s) {
    v16h aK2 = fragA_h(k2b, DD, row0 + lr, s * 32, hi);
    v16h aK = fragA_h(kb_, DD, row0 + lr, s * 32, hi);
    v16h bIV = fragA_f32(iv_f + bb * NS * DD, DD, lr, s * 32, hi);
    v16h bQV = fragA_f32(qiv_f + bb * NS * DD, DD, lr, s * 32, hi);
    g1 = wmma_f16(aK2, bIV, g1);
    g2 = wmma_f16(aK, bQV, g2);
  }
  int slot = lr;
  float cv = c_sum[bb * NS + slot];
  float a[8];
#pragma unroll
  for (int r = 0; r < 8; ++r) {
    float dots = SCALE * (g1[r] - 2.f * g2[r] + cv);
    a[r] = __expf(-dots) + EPSV;   // uniform pi_cl cancels in the ratio
  }
#pragma unroll
  for (int r = 0; r < 8; ++r) {    // normalize over the 16 slots (lanes)
    float tot = a[r];
    tot += __shfl_xor(tot, 1, 32);
    tot += __shfl_xor(tot, 2, 32);
    tot += __shfl_xor(tot, 4, 32);
    tot += __shfl_xor(tot, 8, 32);
    a[r] /= tot;
  }
  float* arow = attn + ((size_t)bb * NS + slot) * NN;
  float part = 0.f;
#pragma unroll
  for (int r = 0; r < 8; ++r) {
    int j = row0 + r + hi * 8;
    arow[j] = a[r];
    part += a[r];
  }
  part += __shfl_xor(part, 16, 32);  // fold the two M halves
  if (hi == 0) partials[((size_t)bb * NTILES + mtile) * NS + slot] = part;
}

// ------------- Per-iter C: deterministic row-sum + normalize ---------
// Writes normalized f32 attn (output path) AND a normalized f16 copy
// that feeds the A fragments of the A@V GEMM directly (no reconvert).
__global__ void k_attn_norm(float* __restrict__ attn, const float* __restrict__ partials,
                            _Float16* __restrict__ attn_h) {
  __shared__ float sm[8];
  int row = blockIdx.x;          // b*ns
  int bb = row / NS, slot = row % NS;
  int t = threadIdx.x;           // 256
  float p = partials[((size_t)bb * NTILES + t) * NS + slot];
  float S = blk_sum(p, sm);
  float invS = 1.f / S;
  float* arow = attn + (size_t)row * NN;
  _Float16* ahrow = attn_h + (size_t)row * NN;
  for (int j = t; j < NN; j += 256) {
    float a = arow[j] * invS;
    arow[j] = a;
    ahrow[j] = (_Float16)a;
  }
}

// ------------- Per-iter D: m = A@V, w = A@V^2 (WMMA, K split) --------
// Both operands fp16 & contiguous; prefetch next K-step (global_prefetch_b8).
__global__ void k_upd(const _Float16* __restrict__ attn_h, const _Float16* __restrict__ vt,
                      float* __restrict__ m_buf, float* __restrict__ w_buf) {
  __shared__ float red[4][2][8][32];
  int lane = threadIdx.x & 31, wv = threadIdx.x >> 5;
  int nt = blockIdx.x;           // 0..7 (d-tiles)
  int bb = blockIdx.y;
  int hi = lane >> 4, lr = lane & 15;
  const _Float16* ab = attn_h + (size_t)bb * NS * NN;
  const _Float16* vb = vt + (size_t)bb * DD * NN;
  v8f accV = {}, accV2 = {};
  for (int s = 0; s < 32; ++s) {             // each wave covers K quarter = 1024
    int k0 = wv * 1024 + s * 32;
    if (s < 31) {
      __builtin_prefetch(ab + (size_t)lr * NN + k0 + 32, 0, 3);
      __builtin_prefetch(vb + (size_t)(nt * 16 + lr) * NN + k0 + 32, 0, 3);
    }
    v16h fa = fragA_h(ab, NN, lr, k0, hi);                 // A rows = slots (f16)
    v16h fb = fragA_h(vb, NN, nt * 16 + lr, k0, hi);       // B from vT (contiguous)
    v16h fb2 = fb * fb;                                    // V^2 on the fly
    accV = wmma_f16(fa, fb, accV);
    accV2 = wmma_f16(fa, fb2, accV2);
  }
#pragma unroll
  for (int r = 0; r < 8; ++r) {
    red[wv][0][r][lane] = accV[r];
    red[wv][1][r][lane] = accV2[r];
  }
  __syncthreads();
  if (wv == 0) {
#pragma unroll
    for (int r = 0; r < 8; ++r) {
      float mv = red[0][0][r][lane] + red[1][0][r][lane] + red[2][0][r][lane] + red[3][0][r][lane];
      float wv2 = red[0][1][r][lane] + red[1][1][r][lane] + red[2][1][r][lane] + red[3][1][r][lane];
      int slot = r + hi * 8;
      int idx = (bb * NS + slot) * DD + nt * 16 + lr;
      m_buf[idx] = mv;
      w_buf[idx] = wv2;
    }
  }
}

// ------------- Generic small GEMM: C = act(A @ W^T + bias) -----------
// A: [M x K] f32 row-major, W: [Nout x K] f32 row-major, C: [M x Nout] f32.
// One wave per (ntile, mtile); K multiple of 32; EXEC all-ones (blockDim 32).
__global__ void k_gemm_tn(const float* __restrict__ A, const float* __restrict__ W,
                          const float* __restrict__ bias, float* __restrict__ C,
                          int Nout, int K, int relu) {
  int lane = threadIdx.x;
  int hi = lane >> 4, lr = lane & 15;
  int ntile = blockIdx.x, mt = blockIdx.y;
  v8f acc = {};
  for (int k0 = 0; k0 < K; k0 += 32) {
    v16h a = fragA_f32(A, K, mt * 16 + lr, k0, hi);
    v16h b = fragA_f32(W, K, ntile * 16 + lr, k0, hi);
    acc = wmma_f16(a, b, acc);
  }
  int col = ntile * 16 + lr;
  float bv = bias[col];
#pragma unroll
  for (int r = 0; r < 8; ++r) {
    int row = mt * 16 + r + hi * 8;
    float v = acc[r] + bv;
    if (relu) v = fmaxf(v, 0.f);
    C[(size_t)row * Nout + col] = v;
  }
}

// ------------- Per-iter E1: GRU gate combine + LN --------------------
__global__ void k_gru_elem(const float* __restrict__ gi, const float* __restrict__ gh,
                           const float* __restrict__ mu_f, const float* __restrict__ g_mu,
                           const float* __restrict__ be_mu, float* __restrict__ u0_buf,
                           float* __restrict__ hb_buf) {
  __shared__ float sm[8];
  int row = blockIdx.x;          // b*ns
  int t = threadIdx.x;           // 128
  const float* gir = gi + (size_t)row * 3 * DD;
  const float* ghr = gh + (size_t)row * 3 * DD;
  float r_ = 1.f / (1.f + __expf(-(gir[t] + ghr[t])));
  float z_ = 1.f / (1.f + __expf(-(gir[DD + t] + ghr[DD + t])));
  float nc = tanhf(gir[2 * DD + t] + r_ * ghr[2 * DD + t]);
  float h = mu_f[row * DD + t];
  float u0 = (1.f - z_) * nc + z_ * h;
  float mean = blk_sum(u0, sm) * (1.f / DD);
  float dv = u0 - mean;
  float var = blk_sum(dv * dv, sm) * (1.f / DD);
  u0_buf[row * DD + t] = u0;
  hb_buf[row * DD + t] = dv * rsqrtf(var + 1e-5f) * g_mu[t] + be_mu[t];
}

// ------------- Per-iter E2: residual + logsig, new slots -------------
__global__ void k_slot_fin(const float* __restrict__ u0_buf, const float* __restrict__ mlpo,
                           const float* __restrict__ m_buf, const float* __restrict__ w_buf,
                           float* __restrict__ slots_out) {
  int row = blockIdx.x;          // b*ns
  int t = threadIdx.x;           // 128
  float u = u0_buf[row * DD + t] + mlpo[row * DD + t];
  float mm = m_buf[row * DD + t];
  float ww = w_buf[row * DD + t];
  // ls = 0.5*log(A@V^2 - 2*u*m + u^2 + EPS)   (sum_j attn = 1)
  float arg = ww - 2.f * u * mm + u * u + EPSV;
  slots_out[row * 2 * DD + t] = u;
  slots_out[row * 2 * DD + DD + t] = 0.5f * __logf(fmaxf(arg, 1e-30f));
}

extern "C" void kernel_launch(void* const* d_in, const int* in_sizes, int n_in,
                              void* d_out, int out_size, void* d_ws, size_t ws_size,
                              hipStream_t stream) {
  (void)in_sizes; (void)n_in; (void)out_size; (void)ws_size;
  const float* slots  = (const float*)d_in[0];
  const float* inputs = (const float*)d_in[1];
  const float* Wq     = (const float*)d_in[2];
  const float* Wk     = (const float*)d_in[3];
  const float* Wv     = (const float*)d_in[4];
  const float* W_ih   = (const float*)d_in[5];
  const float* W_hh   = (const float*)d_in[6];
  const float* b_ih   = (const float*)d_in[7];
  const float* b_hh   = (const float*)d_in[8];
  const float* mu_W1  = (const float*)d_in[9];
  const float* mu_b1  = (const float*)d_in[10];
  const float* mu_W2  = (const float*)d_in[11];
  const float* mu_b2  = (const float*)d_in[12];
  const float* g_in   = (const float*)d_in[13];
  const float* be_in  = (const float*)d_in[14];
  const float* g_sl   = (const float*)d_in[15];
  const float* be_sl  = (const float*)d_in[16];
  const float* g_mu   = (const float*)d_in[17];
  const float* be_mu  = (const float*)d_in[18];
  const float* out_W1 = (const float*)d_in[19];
  const float* out_b1 = (const float*)d_in[20];
  const float* out_W2 = (const float*)d_in[21];
  const float* out_b2 = (const float*)d_in[22];
  float* out = (float*)d_out;

  char* ws = (char*)d_ws;
  size_t off = 0;
  auto carve = [&](size_t bytes) -> char* {
    char* p = ws + off;
    off = (off + bytes + 255) & ~(size_t)255;
    return p;
  };
  const int NR = BB * NS;  // 64 slot rows total
  _Float16* x_h    = (_Float16*)carve((size_t)BB * NN * DD * 2);
  _Float16* k_h    = (_Float16*)carve((size_t)BB * NN * DD * 2);
  _Float16* k2_h   = (_Float16*)carve((size_t)BB * NN * DD * 2);
  _Float16* vt_h   = (_Float16*)carve((size_t)BB * NN * DD * 2);
  _Float16* attn_h = (_Float16*)carve((size_t)BB * NS * NN * 2);
  float* iv_f      = (float*)carve((size_t)NR * DD * 4);
  float* qiv_f     = (float*)carve((size_t)NR * DD * 4);
  float* c_sum     = (float*)carve((size_t)NR * 4);
  float* mu_f      = (float*)carve((size_t)NR * DD * 4);
  float* attn      = (float*)carve((size_t)BB * NS * NN * 4);
  float* partials  = (float*)carve((size_t)BB * NTILES * NS * 4);
  float* m_buf     = (float*)carve((size_t)NR * DD * 4);
  float* w_buf     = (float*)carve((size_t)NR * DD * 4);
  float* slots_cur = (float*)carve((size_t)NR * 2 * DD * 4);
  float* gi_buf    = (float*)carve((size_t)NR * 3 * DD * 4);
  float* gh_buf    = (float*)carve((size_t)NR * 3 * DD * 4);
  float* u0_buf    = (float*)carve((size_t)NR * DD * 4);
  float* hb_buf    = (float*)carve((size_t)NR * DD * 4);
  float* hid_buf   = (float*)carve((size_t)NR * HID * 4);
  float* mlpo_buf  = (float*)carve((size_t)NR * DD * 4);
  float* hido_buf  = (float*)carve((size_t)NR * 2 * HID * 4);

  // Stage 0/1: LN inputs, project to k/v (k,k^2 row-major; v transposed), fp16
  k_ln_in<<<BB * NN, 128, 0, stream>>>(inputs, g_in, be_in, x_h);
  k_proj<<<(BB * NN / 16) / 4, 128, 0, stream>>>(x_h, Wk, Wv, k_h, k2_h, vt_h);

  const float* cur_slots = slots;
  for (int it = 0; it < 4; ++it) {   // ITERS=3 + final stop_gradient step
    k_slot_pre<<<NR, 128, 0, stream>>>(cur_slots, g_sl, be_sl, Wq,
                                       iv_f, qiv_f, c_sum, mu_f);
    k_dots<<<dim3(NTILES / 4, BB), 128, 0, stream>>>(k_h, k2_h, iv_f, qiv_f,
                                                     c_sum, attn, partials);
    k_attn_norm<<<NR, 256, 0, stream>>>(attn, partials, attn_h);
    k_upd<<<dim3(8, BB), 128, 0, stream>>>(attn_h, vt_h, m_buf, w_buf);
    // GRU gates via WMMA: gi = m@W_ih^T + b_ih ; gh = mu@W_hh^T + b_hh
    k_gemm_tn<<<dim3(3 * DD / 16, NR / 16), 32, 0, stream>>>(m_buf, W_ih, b_ih, gi_buf,
                                                             3 * DD, DD, 0);
    k_gemm_tn<<<dim3(3 * DD / 16, NR / 16), 32, 0, stream>>>(mu_f, W_hh, b_hh, gh_buf,
                                                             3 * DD, DD, 0);
    k_gru_elem<<<NR, 128, 0, stream>>>(gi_buf, gh_buf, mu_f, g_mu, be_mu, u0_buf, hb_buf);
    // mu-MLP via WMMA: relu(h@W1^T+b1)@W2^T+b2
    k_gemm_tn<<<dim3(HID / 16, NR / 16), 32, 0, stream>>>(hb_buf, mu_W1, mu_b1, hid_buf,
                                                          HID, DD, 1);
    k_gemm_tn<<<dim3(DD / 16, NR / 16), 32, 0, stream>>>(hid_buf, mu_W2, mu_b2, mlpo_buf,
                                                         DD, HID, 0);
    k_slot_fin<<<NR, 128, 0, stream>>>(u0_buf, mlpo_buf, m_buf, w_buf, slots_cur);
    cur_slots = slots_cur;
  }
  // Output MLP via WMMA: relu(slots@out_W1^T+b1)@out_W2^T+b2 -> d_out[0:8192]
  k_gemm_tn<<<dim3(2 * HID / 16, NR / 16), 32, 0, stream>>>(slots_cur, out_W1, out_b1,
                                                            hido_buf, 2 * HID, 2 * DD, 1);
  k_gemm_tn<<<dim3(DD / 16, NR / 16), 32, 0, stream>>>(hido_buf, out_W2, out_b2, out,
                                                       DD, 2 * HID, 0);
  hipMemcpyAsync(out + (size_t)NR * DD, attn, (size_t)BB * NS * NN * sizeof(float),
                 hipMemcpyDeviceToDevice, stream);
}
